// LocalMessagePassing_66073776881929
// MI455X (gfx1250) — matrix operand order repaired and checked by reading, not compiled, not run on hardware
//
#include <hip/hip_runtime.h>
#include <hip/hip_bf16.h>

// ---------------------------------------------------------------------------
// MI455X (gfx1250) implementation of the routed message-passing charge model.
//
// - GEMMs: f16 operands, f32 accumulation via v_wmma_f32_16x16x32_f16.
//   Activations (16MB each) are L2-resident -> matrix-pipe bound, not HBM.
// - Species routing: counting sort so every 16-row M-tile is species-uniform;
//   B operands pre-packed in the ISA B-fragment lane layout (2x b128/lane).
// - B fragments are loaded once per (nt,ks) and shared by both M-tiles of the
//   block (tiles are species-sorted, so s0==s1 almost always; uniform branch).
// - Edge scatter: L2-resident global_atomic_add_f32.
// - k_final stages the 4KB projection weights into LDS with
//   global_load_async_to_lds_b128 + s_wait_asynccnt (CDNA5 async path).
// ---------------------------------------------------------------------------

typedef _Float16 h16;
typedef _Float16 v8h  __attribute__((ext_vector_type(8)));
typedef _Float16 v16h __attribute__((ext_vector_type(16)));
typedef float    v8f  __attribute__((ext_vector_type(8)));

#define NBATCH  256
#define NA      128
#define NATOMS  32768      // NBATCH*NA
#define FDIM    128
#define HDIM    256
#define MODIM   128
#define NODIM   128
#define PEDGE   524288
#define NSPEC   4
#define NT_MAX  2052       // sum ceil(count_s/16) <= N/16 + S
#define CUTOFF_F 5.2f

// ---------------------------------------------------------------------------
// WMMA helpers
// ---------------------------------------------------------------------------
__device__ __forceinline__ v8f wmma_f16(v16h a, v16h b, v8f c) {
  // D = A(16x32 f16) * B(32x16 f16) + C(16x16 f32)
  return __builtin_amdgcn_wmma_f32_16x16x32_f16(
      /*neg_a=*/false, a, /*neg_b=*/false, b,
      /*c_mod=*/(short)0, c, /*reuse_a=*/false, /*reuse_b=*/false);
}

// A fragment from row-major f16 LDS tile.
// ISA A layout (16-bit, 16x32): lane<16 holds row M=lane, K = base+{0..7,16..23};
// lane>=16 holds M=lane-16, K = base+{8..15, 24..31}.
__device__ __forceinline__ v16h load_a(const h16* rowbase, int stride, int kbase, int lane) {
  int m  = lane & 15;
  int hf = lane >> 4;
  const h16* p = rowbase + m * stride + kbase + hf * 8;
  v8h lo = *(const v8h*)(p);
  v8h hi = *(const v8h*)(p + 16);
  v16h a;
#pragma unroll
  for (int t = 0; t < 8; ++t) { a[t] = lo[t]; a[8 + t] = hi[t]; }
  return a;
}

// B fragment: weights pre-packed in the ISA B lane layout; each lane reads
// 32 contiguous bytes (two global_load_b128, L2-hot).
__device__ __forceinline__ v16h load_b(const unsigned int* __restrict__ Wp,
                                       int s, int KT, int NT, int kt, int nt, int lane) {
  const uint4* p = (const uint4*)(Wp + ((((long)s * KT + kt) * NT + nt) * 32 + lane) * 8);
  union { uint4 u[2]; v16h h; } cv;
  cv.u[0] = p[0];
  cv.u[1] = p[1];
  return cv.h;
}

// ---------------------------------------------------------------------------
// Small prep kernels
// ---------------------------------------------------------------------------
__global__ void k_zerof(float* p, long n) {
  long i = (long)blockIdx.x * blockDim.x + threadIdx.x;
  long stride = (long)gridDim.x * blockDim.x;
  for (; i < n; i += stride) p[i] = 0.0f;
}

__global__ void k_edge(const float* __restrict__ dist,
                       const float* __restrict__ prefactor,
                       const float* __restrict__ factor,
                       float* __restrict__ ew) {
  int p = blockIdx.x * 256 + threadIdx.x;
  if (p >= PEDGE) return;
  float d  = dist[p];
  float pf = prefactor[0], fc = factor[0];
  float decay = pf * pf * expf(-fc * fc * d);
  float cut = (d < CUTOFF_F) ? (0.5f * cosf(3.14159265358979323846f * d / CUTOFF_F) + 0.5f) : 0.0f;
  ew[p] = decay * cut;
}

__global__ void k_hist(const int* __restrict__ sp, int* counts) {
  int n = blockIdx.x * 256 + threadIdx.x;
  if (n < NATOMS) atomicAdd(&counts[sp[n]], 1);
}

__global__ void k_offsets(const int* __restrict__ counts, int* segOff, int* cursor, int* tileBase) {
  if (threadIdx.x == 0 && blockIdx.x == 0) {
    int off = 0, tb = 0;
    for (int s = 0; s < NSPEC; ++s) {
      segOff[s] = off; cursor[s] = off; tileBase[s] = tb;
      off += counts[s];
      tb  += (counts[s] + 15) >> 4;
    }
    tileBase[NSPEC] = tb;
  }
}

__global__ void k_perm(const int* __restrict__ sp, int* cursor, int* __restrict__ perm) {
  int n = blockIdx.x * 256 + threadIdx.x;
  if (n < NATOMS) {
    int pos = atomicAdd(&cursor[sp[n]], 1);
    perm[pos] = n;
  }
}

// Build per-tile routing table: each 16-row M-tile is species-uniform.
__global__ void k_tiles(const int* __restrict__ counts, const int* __restrict__ segOff,
                        const int* __restrict__ tileBase, const int* __restrict__ perm,
                        int* __restrict__ tileSpec, int* __restrict__ tileRows,
                        int* __restrict__ rowAtom) {
  for (int t = threadIdx.x; t < NT_MAX; t += 256) {
    int s = -1;
    for (int q = 0; q < NSPEC; ++q)
      if (t >= tileBase[q] && t < tileBase[q + 1]) s = q;
    if (s < 0) {
      tileSpec[t] = 0; tileRows[t] = 0;
      for (int m = 0; m < 16; ++m) rowAtom[t * 16 + m] = 0;
    } else {
      int local = t - tileBase[s];
      int cs = counts[s];
      int rows = cs - local * 16; if (rows > 16) rows = 16;
      tileSpec[t] = s; tileRows[t] = rows;
      for (int m = 0; m < 16; ++m) {
        int r = local * 16 + m;
        rowAtom[t * 16 + m] = (r < cs) ? perm[segOff[s] + r]
                                       : (cs > 0 ? perm[segOff[s]] : 0);
      }
    }
  }
}

// Repack f32 weights (S,K,H) into f16 WMMA-B lane layout:
// dst[((s*KT+kt)*NT+nt)*32 + lane][j] packs B[k0,n]|B[k0+1,n]<<16 with
// n = nt*16 + lane%16, k0 = kt*32 + (j/4)*16 + (lane/16)*8 + (j%4)*2.
__global__ void k_packw(const float* __restrict__ src, unsigned int* __restrict__ dst,
                        int K, int H) {
  int  KT = K >> 5, NT = H >> 4;
  long total = (long)NSPEC * K * H / 2;
  long id = (long)blockIdx.x * 256 + threadIdx.x;
  if (id >= total) return;
  int j    = (int)(id & 7);
  int lane = (int)((id >> 3) & 31);
  long rest = id >> 8;
  int nt = (int)(rest % NT); rest /= NT;
  int kt = (int)(rest % KT);
  int s  = (int)(rest / KT);
  int n  = nt * 16 + (lane & 15);
  int k0 = kt * 32 + (j >> 2) * 16 + (lane >> 4) * 8 + (j & 3) * 2;
  const float* base = src + (long)s * K * H;
  union { h16 h[2]; unsigned int u; } cv;
  cv.h[0] = (h16)base[(long)k0 * H + n];
  cv.h[1] = (h16)base[(long)(k0 + 1) * H + n];
  dst[id] = cv.u;
}

// ---------------------------------------------------------------------------
// Fused routed 2-layer MLP: out = celu(X @ W1[s] + b1[s]) @ W2[s] + b2[s]
// Block = 2 species-uniform M-tiles (32 atoms), 8 wave32s. B fragments are
// shared by both M-tiles when species match (the common, sorted case).
// ---------------------------------------------------------------------------
template <int K>
__global__ __launch_bounds__(256) void k_mlp(
    const float* __restrict__ Xa, const float* __restrict__ Xb,
    const unsigned int* __restrict__ W1p, const float* __restrict__ b1,
    const unsigned int* __restrict__ W2p, const float* __restrict__ b2,
    float* __restrict__ out,
    const int* __restrict__ tileSpec, const int* __restrict__ tileRows,
    const int* __restrict__ rowAtom) {
  constexpr int XS = K + 8;        // padded LDS row stride (bank spread)
  constexpr int HS = HDIM + 8;
  __shared__ __align__(16) h16 Xl[32 * XS];
  __shared__ __align__(16) h16 Hl[32 * HS];
  __shared__ int sAtom[32];
  __shared__ int sSpec[2];
  __shared__ int sRows[2];

  int tid = threadIdx.x;
  int tbase = blockIdx.x * 2;
  if (tid < 2) { sSpec[tid] = tileSpec[tbase + tid]; sRows[tid] = tileRows[tbase + tid]; }
  if (tid < 32) sAtom[tid] = rowAtom[blockIdx.x * 32 + tid];
  __syncthreads();
  if (sRows[0] + sRows[1] == 0) return;   // block-uniform

  // Stage gathered X rows -> LDS, f32 -> f16 convert on the fly.
  constexpr int C4 = K / 4;
#pragma unroll
  for (int it = 0; it < (32 * C4) / 256; ++it) {
    int id  = it * 256 + tid;
    int row = id / C4, c4 = id % C4;
    int col = c4 * 4;
    long atom = sAtom[row];
    float4 v;
    if (K == 256 && col >= 128) v = *(const float4*)(Xb + atom * 128 + (col - 128));
    else                        v = *(const float4*)(Xa + atom * 128 + col);
    h16* d = &Xl[row * XS + col];
    d[0] = (h16)v.x; d[1] = (h16)v.y; d[2] = (h16)v.z; d[3] = (h16)v.w;
  }
  __syncthreads();

  int wv = tid >> 5, lane = tid & 31;
  int n_in = lane & 15, hf = lane >> 4;
  constexpr int KT1 = K / 32;
  const int s0 = sSpec[0], s1 = sSpec[1];
  const bool same = (s0 == s1);            // block-uniform

  // ---- layer 1: H = celu(X @ W1 + b1), kept in LDS as f16 ----
#pragma unroll
  for (int nti = 0; nti < 2; ++nti) {
    int nt = wv * 2 + nti;                 // 16 N-tiles of H=256
    v8f acc0 = {0.f, 0.f, 0.f, 0.f, 0.f, 0.f, 0.f, 0.f};
    v8f acc1 = {0.f, 0.f, 0.f, 0.f, 0.f, 0.f, 0.f, 0.f};
    if (same) {
#pragma unroll
      for (int ks = 0; ks < KT1; ++ks) {
        v16h b  = load_b(W1p, s0, KT1, HDIM / 16, ks, nt, lane);
        v16h a0 = load_a(Xl, XS, ks * 32, lane);
        v16h a1 = load_a(Xl + 16 * XS, XS, ks * 32, lane);
        acc0 = wmma_f16(a0, b, acc0);
        acc1 = wmma_f16(a1, b, acc1);
      }
    } else {
#pragma unroll
      for (int ks = 0; ks < KT1; ++ks) {
        v16h a0 = load_a(Xl, XS, ks * 32, lane);
        acc0 = wmma_f16(a0, load_b(W1p, s0, KT1, HDIM / 16, ks, nt, lane), acc0);
        v16h a1 = load_a(Xl + 16 * XS, XS, ks * 32, lane);
        acc1 = wmma_f16(a1, load_b(W1p, s1, KT1, HDIM / 16, ks, nt, lane), acc1);
      }
    }
    int col = nt * 16 + n_in;
    float bias0 = b1[s0 * HDIM + col];
    float bias1 = b1[s1 * HDIM + col];
#pragma unroll
    for (int j = 0; j < 8; ++j) {
      float x0 = acc0[j] + bias0;
      x0 = (x0 > 0.0f) ? x0 : expm1f(x0);            // celu(alpha=1)
      Hl[(j + 8 * hf) * HS + col] = (h16)x0;
      float x1 = acc1[j] + bias1;
      x1 = (x1 > 0.0f) ? x1 : expm1f(x1);
      Hl[(16 + j + 8 * hf) * HS + col] = (h16)x1;
    }
  }
  __syncthreads();

  // ---- layer 2: out = H @ W2 + b2 (scatter to original atom ids) ----
  {
    int nt = wv;                           // 8 N-tiles of MO=128
    int col = nt * 16 + n_in;
    v8f acc0 = {0.f, 0.f, 0.f, 0.f, 0.f, 0.f, 0.f, 0.f};
    v8f acc1 = {0.f, 0.f, 0.f, 0.f, 0.f, 0.f, 0.f, 0.f};
    if (same) {
#pragma unroll
      for (int ks = 0; ks < HDIM / 32; ++ks) {
        v16h b  = load_b(W2p, s0, HDIM / 32, MODIM / 16, ks, nt, lane);
        v16h a0 = load_a(Hl, HS, ks * 32, lane);
        v16h a1 = load_a(Hl + 16 * HS, HS, ks * 32, lane);
        acc0 = wmma_f16(a0, b, acc0);
        acc1 = wmma_f16(a1, b, acc1);
      }
    } else {
#pragma unroll
      for (int ks = 0; ks < HDIM / 32; ++ks) {
        v16h a0 = load_a(Hl, HS, ks * 32, lane);
        acc0 = wmma_f16(a0, load_b(W2p, s0, HDIM / 32, MODIM / 16, ks, nt, lane), acc0);
        v16h a1 = load_a(Hl + 16 * HS, HS, ks * 32, lane);
        acc1 = wmma_f16(a1, load_b(W2p, s1, HDIM / 32, MODIM / 16, ks, nt, lane), acc1);
      }
    }
    // tile 0
    {
      int rows = sRows[0], rb = 8 * hf;
      float bias = b2[s0 * MODIM + col];
      if (rows == 16) {                    // fast path (block-uniform)
#pragma unroll
        for (int j = 0; j < 8; ++j) {
          long atom = sAtom[rb + j];
          out[atom * MODIM + col] = acc0[j] + bias;
        }
      } else {
#pragma unroll
        for (int j = 0; j < 8; ++j)
          if (rb + j < rows) {
            long atom = sAtom[rb + j];
            out[atom * MODIM + col] = acc0[j] + bias;
          }
      }
    }
    // tile 1
    {
      int rows = sRows[1], rb = 8 * hf;
      float bias = b2[s1 * MODIM + col];
      if (rows == 16) {
#pragma unroll
        for (int j = 0; j < 8; ++j) {
          long atom = sAtom[16 + rb + j];
          out[atom * MODIM + col] = acc1[j] + bias;
        }
      } else {
#pragma unroll
        for (int j = 0; j < 8; ++j)
          if (rb + j < rows) {
            long atom = sAtom[16 + rb + j];
            out[atom * MODIM + col] = acc1[j] + bias;
          }
      }
    }
  }
}

// ---------------------------------------------------------------------------
// Routed single linear: out = X @ W[s] + b[s]   (K = MO = 128, NOUT = 128)
// ---------------------------------------------------------------------------
__global__ __launch_bounds__(256) void k_lin(
    const float* __restrict__ X,
    const unsigned int* __restrict__ Wp, const float* __restrict__ bias_,
    float* __restrict__ out,
    const int* __restrict__ tileSpec, const int* __restrict__ tileRows,
    const int* __restrict__ rowAtom) {
  constexpr int K = MODIM;
  constexpr int XS = K + 8;
  __shared__ __align__(16) h16 Xl[32 * XS];
  __shared__ int sAtom[32];
  __shared__ int sSpec[2];
  __shared__ int sRows[2];

  int tid = threadIdx.x;
  int tbase = blockIdx.x * 2;
  if (tid < 2) { sSpec[tid] = tileSpec[tbase + tid]; sRows[tid] = tileRows[tbase + tid]; }
  if (tid < 32) sAtom[tid] = rowAtom[blockIdx.x * 32 + tid];
  __syncthreads();
  if (sRows[0] + sRows[1] == 0) return;

  constexpr int C4 = K / 4;
#pragma unroll
  for (int it = 0; it < (32 * C4) / 256; ++it) {
    int id  = it * 256 + tid;
    int row = id / C4, c4 = id % C4;
    int col = c4 * 4;
    long atom = sAtom[row];
    float4 v = *(const float4*)(X + atom * 128 + col);
    h16* d = &Xl[row * XS + col];
    d[0] = (h16)v.x; d[1] = (h16)v.y; d[2] = (h16)v.z; d[3] = (h16)v.w;
  }
  __syncthreads();

  int wv = tid >> 5, lane = tid & 31;
  int n_in = lane & 15, hf = lane >> 4;
  const int s0 = sSpec[0], s1 = sSpec[1];
  const bool same = (s0 == s1);
  int nt = wv;                             // 8 N-tiles of NO=128
  int col = nt * 16 + n_in;
  v8f acc0 = {0.f, 0.f, 0.f, 0.f, 0.f, 0.f, 0.f, 0.f};
  v8f acc1 = {0.f, 0.f, 0.f, 0.f, 0.f, 0.f, 0.f, 0.f};
  if (same) {
#pragma unroll
    for (int ks = 0; ks < K / 32; ++ks) {
      v16h b  = load_b(Wp, s0, K / 32, NODIM / 16, ks, nt, lane);
      v16h a0 = load_a(Xl, XS, ks * 32, lane);
      v16h a1 = load_a(Xl + 16 * XS, XS, ks * 32, lane);
      acc0 = wmma_f16(a0, b, acc0);
      acc1 = wmma_f16(a1, b, acc1);
    }
  } else {
#pragma unroll
    for (int ks = 0; ks < K / 32; ++ks) {
      v16h a0 = load_a(Xl, XS, ks * 32, lane);
      acc0 = wmma_f16(a0, load_b(Wp, s0, K / 32, NODIM / 16, ks, nt, lane), acc0);
      v16h a1 = load_a(Xl + 16 * XS, XS, ks * 32, lane);
      acc1 = wmma_f16(a1, load_b(Wp, s1, K / 32, NODIM / 16, ks, nt, lane), acc1);
    }
  }
  {
    int rows = sRows[0], rb = 8 * hf;
    float bias = bias_[s0 * NODIM + col];
    if (rows == 16) {
#pragma unroll
      for (int j = 0; j < 8; ++j) {
        long atom = sAtom[rb + j];
        out[atom * NODIM + col] = acc0[j] + bias;
      }
    } else {
#pragma unroll
      for (int j = 0; j < 8; ++j)
        if (rb + j < rows) {
          long atom = sAtom[rb + j];
          out[atom * NODIM + col] = acc0[j] + bias;
        }
    }
  }
  {
    int rows = sRows[1], rb = 8 * hf;
    float bias = bias_[s1 * NODIM + col];
    if (rows == 16) {
#pragma unroll
      for (int j = 0; j < 8; ++j) {
        long atom = sAtom[16 + rb + j];
        out[atom * NODIM + col] = acc1[j] + bias;
      }
    } else {
#pragma unroll
      for (int j = 0; j < 8; ++j)
        if (rb + j < rows) {
          long atom = sAtom[16 + rb + j];
          out[atom * NODIM + col] = acc1[j] + bias;
        }
    }
  }
}

// ---------------------------------------------------------------------------
// Symmetric edge scatter: merged[i] += w*neigh[j]; merged[j] += w*neigh[i].
// One wave per edge, lane owns 4 channels; L2-resident f32 atomics.
// ---------------------------------------------------------------------------
__global__ __launch_bounds__(256) void k_scatter(
    const int* __restrict__ idx, const float* __restrict__ ew,
    const float* __restrict__ neigh, float* __restrict__ merged) {
  int e = blockIdx.x * 8 + (threadIdx.x >> 5);
  int lane = threadIdx.x & 31;
  if (e >= PEDGE) return;
  long i = idx[e];
  long j = idx[PEDGE + e];
  float w = ew[e];
  int c = lane * 4;
  float4 nj = *(const float4*)(neigh + j * 128 + c);
  float4 ni = *(const float4*)(neigh + i * 128 + c);
  float* mi = merged + i * 128 + c;
  float* mj = merged + j * 128 + c;
  atomicAdd(mi + 0, nj.x * w); atomicAdd(mi + 1, nj.y * w);
  atomicAdd(mi + 2, nj.z * w); atomicAdd(mi + 3, nj.w * w);
  atomicAdd(mj + 0, ni.x * w); atomicAdd(mj + 1, ni.y * w);
  atomicAdd(mj + 2, ni.z * w); atomicAdd(mj + 3, ni.w * w);
}

// ---------------------------------------------------------------------------
// Final: precharge GEMV (K=256, NOUT=1) + per-batch renormalization.
// Projection weights (4 species x 1KB) staged to LDS with the CDNA5 async
// LDS-DMA path. One block per batch (128 atoms).
// Outputs: [species | charges | precharges].
// ---------------------------------------------------------------------------
__global__ __launch_bounds__(128) void k_final(
    const int* __restrict__ species,
    const float* __restrict__ A2, const float* __restrict__ MG2,
    const float* __restrict__ fw, const float* __restrict__ fb,
    const float* __restrict__ tc, float* __restrict__ out) {
  __shared__ __align__(16) float wsh[NSPEC * (MODIM + NODIM)];  // 4KB
  __shared__ float red[128];
  int b = blockIdx.x, a = threadIdx.x;

  // Async-stage all species' projection weights: 256 x 16B chunks, 2 per thread.
#pragma unroll
  for (int it = 0; it < 2; ++it) {
    int chunk = it * 128 + a;
    unsigned lds_addr = (unsigned)(size_t)(&wsh[chunk * 4]);
    const float* gsrc = fw + chunk * 4;
    asm volatile("global_load_async_to_lds_b128 %0, %1, off"
                 :: "v"(lds_addr), "v"(gsrc) : "memory");
  }
  asm volatile("s_wait_asynccnt 0" ::: "memory");
  __syncthreads();

  long n = (long)b * NA + a;
  int s = species[n];
  const float* w = wsh + s * (MODIM + NODIM);
  float acc = 0.0f;
#pragma unroll 4
  for (int c = 0; c < 128; c += 4) {
    float4 xa = *(const float4*)(A2 + n * 128 + c);
    float4 xb = *(const float4*)(MG2 + n * 128 + c);
    float4 wa = *(const float4*)(w + c);
    float4 wb = *(const float4*)(w + 128 + c);
    acc += xa.x * wa.x + xa.y * wa.y + xa.z * wa.z + xa.w * wa.w;
    acc += xb.x * wb.x + xb.y * wb.y + xb.z * wb.z + xb.w * wb.w;
  }
  float pre = acc + fb[s];
  red[a] = pre;
  __syncthreads();
  for (int st = 64; st > 0; st >>= 1) {
    if (a < st) red[a] += red[a + st];
    __syncthreads();
  }
  float total = red[0];
  float charge = pre + (tc[b] - total) * (1.0f / (float)NA);  // dummy_mask all true
  out[n]              = (float)s;     // species echoed as float in concat output
  out[NATOMS + n]     = charge;
  out[2 * NATOMS + n] = pre;
}

// ---------------------------------------------------------------------------
// Host-side orchestration (graph-capture safe: stream-ordered launches only)
// ---------------------------------------------------------------------------
extern "C" void kernel_launch(void* const* d_in, const int* in_sizes, int n_in,
                              void* d_out, int out_size, void* d_ws, size_t ws_size,
                              hipStream_t stream) {
  const int*   species   = (const int*)  d_in[0];
  const float* in_feats  = (const float*)d_in[1];   // (N, 128) contiguous
  const int*   aidx      = (const int*)  d_in[2];   // (2, P)
  const float* dist      = (const float*)d_in[3];
  const float* tcharges  = (const float*)d_in[4];
  const float* p0_w1 = (const float*)d_in[5];
  const float* p0_b1 = (const float*)d_in[6];
  const float* p0_w2 = (const float*)d_in[7];
  const float* p0_b2 = (const float*)d_in[8];
  const float* n0_w  = (const float*)d_in[9];
  const float* n0_b  = (const float*)d_in[10];
  const float* p1_w1 = (const float*)d_in[11];
  const float* p1_b1 = (const float*)d_in[12];
  const float* p1_w2 = (const float*)d_in[13];
  const float* p1_b2 = (const float*)d_in[14];
  const float* n1_w  = (const float*)d_in[15];
  const float* n1_b  = (const float*)d_in[16];
  const float* f_w   = (const float*)d_in[17];
  const float* f_b   = (const float*)d_in[18];
  const float* prefactor = (const float*)d_in[19];
  const float* factor    = (const float*)d_in[20];

  char* ws = (char*)d_ws;
  size_t o = 0;
  auto alloc = [&](size_t bytes) -> char* {
    char* r = ws + o;
    o = (o + bytes + 255) & ~(size_t)255;
    return r;
  };
  float* ew   = (float*)alloc((size_t)PEDGE * 4);
  float* A1   = (float*)alloc((size_t)NATOMS * 128 * 4);   // pass-1 internal
  float* NBuf = (float*)alloc((size_t)NATOMS * 128 * 4);   // neigh (reused)
  float* MG1  = (float*)alloc((size_t)NATOMS * 128 * 4);   // pass-1 merged
  float* A2   = (float*)alloc((size_t)NATOMS * 128 * 4);   // pass-2 internal
  float* MG2  = (float*)alloc((size_t)NATOMS * 128 * 4);   // pass-2 merged
  int*   perm = (int*)  alloc((size_t)NATOMS * 4);
  int*   meta = (int*)  alloc(32 * 4);
  int* counts   = meta;
  int* cursor   = meta + 4;
  int* segOff   = meta + 8;
  int* tileBase = meta + 12;
  int* tileSpec = (int*)alloc((size_t)NT_MAX * 4);
  int* tileRows = (int*)alloc((size_t)NT_MAX * 4);
  int* rowAtom  = (int*)alloc((size_t)NT_MAX * 16 * 4);
  unsigned int* W1a = (unsigned int*)alloc((size_t)NSPEC * FDIM  * HDIM  * 2);
  unsigned int* W2a = (unsigned int*)alloc((size_t)NSPEC * HDIM  * MODIM * 2);
  unsigned int* NWa = (unsigned int*)alloc((size_t)NSPEC * MODIM * NODIM * 2);
  unsigned int* W1b = (unsigned int*)alloc((size_t)NSPEC * (MODIM + NODIM) * HDIM * 2);
  unsigned int* W2b = (unsigned int*)alloc((size_t)NSPEC * HDIM  * MODIM * 2);
  unsigned int* NWb = (unsigned int*)alloc((size_t)NSPEC * MODIM * NODIM * 2);

  // --- zero scatter targets + sort metadata ---
  k_zerof<<<1024, 256, 0, stream>>>(MG1, (long)NATOMS * 128);
  k_zerof<<<1024, 256, 0, stream>>>(MG2, (long)NATOMS * 128);
  k_zerof<<<1, 32, 0, stream>>>((float*)meta, 8);   // counts + cursor

  // --- edge weights ---
  k_edge<<<PEDGE / 256, 256, 0, stream>>>(dist, prefactor, factor, ew);

  // --- species counting sort + tile table ---
  k_hist<<<NATOMS / 256, 256, 0, stream>>>(species, counts);
  k_offsets<<<1, 32, 0, stream>>>(counts, segOff, cursor, tileBase);
  k_perm<<<NATOMS / 256, 256, 0, stream>>>(species, cursor, perm);
  k_tiles<<<1, 256, 0, stream>>>(counts, segOff, tileBase, perm, tileSpec, tileRows, rowAtom);

  // --- pack weights into WMMA B-fragment lane layout (f32 -> f16) ---
  auto packw = [&](const float* src, unsigned int* dst, int K, int H) {
    long total = (long)NSPEC * K * H / 2;
    k_packw<<<(int)((total + 255) / 256), 256, 0, stream>>>(src, dst, K, H);
  };
  packw(p0_w1, W1a, FDIM, HDIM);
  packw(p0_w2, W2a, HDIM, MODIM);
  packw(n0_w,  NWa, MODIM, NODIM);
  packw(p1_w1, W1b, MODIM + NODIM, HDIM);
  packw(p1_w2, W2b, HDIM, MODIM);
  packw(n1_w,  NWb, MODIM, NODIM);

  const int GEMM_BLOCKS = NT_MAX / 2;   // 2 M-tiles per block

  // --- pass 1 ---
  k_mlp<128><<<GEMM_BLOCKS, 256, 0, stream>>>(in_feats, in_feats, W1a, p0_b1, W2a, p0_b2,
                                              A1, tileSpec, tileRows, rowAtom);
  k_lin<<<GEMM_BLOCKS, 256, 0, stream>>>(A1, NWa, n0_b, NBuf, tileSpec, tileRows, rowAtom);
  k_scatter<<<PEDGE / 8, 256, 0, stream>>>(aidx, ew, NBuf, MG1);

  // --- pass 2 (features = [A1 | MG1], K = 256) ---
  k_mlp<256><<<GEMM_BLOCKS, 256, 0, stream>>>(A1, MG1, W1b, p1_b1, W2b, p1_b2,
                                              A2, tileSpec, tileRows, rowAtom);
  k_lin<<<GEMM_BLOCKS, 256, 0, stream>>>(A2, NWb, n1_b, NBuf, tileSpec, tileRows, rowAtom);
  k_scatter<<<PEDGE / 8, 256, 0, stream>>>(aidx, ew, NBuf, MG2);

  // --- final charges ---
  k_final<<<NBATCH, 128, 0, stream>>>(species, A2, MG2, f_w, f_b, tcharges, (float*)d_out);
}
